// AlgebraicTransformerLM_48756468744652
// MI455X (gfx1250) — compile-verified, matrix operand order checked by reading, not run on hardware
//
#include <hip/hip_runtime.h>
#include <hip/hip_bf16.h>
#include <cstdint>
#include <cstddef>

// ---------------- model dims ----------------
static constexpr int   BB  = 2;
static constexpr int   TT  = 1024;
static constexpr int   VV  = 32000;
static constexpr int   DD  = 1024;
static constexpr int   HH  = 16;
static constexpr int   LL  = 4;
static constexpr int   FF  = 2730;
static constexpr int   DHH = 64;          // head dim
static constexpr int   FPAD= 2752;        // F padded to multiple of 64 (WMMA K)
static constexpr int   GPAD= 5504;        // 2F padded to multiple of 64 (GEMM N)
static constexpr int   MMr = BB * TT;     // 2048 token rows
#define EPSV 1e-6f
#define NEGV -10000.0f

typedef __bf16 bf16_t;
typedef __attribute__((ext_vector_type(16))) __bf16 v16bf;
typedef __attribute__((ext_vector_type(8)))  __bf16 v8bf;
typedef __attribute__((ext_vector_type(8)))  float  v8f;

__device__ __forceinline__ bf16_t f2bf(float f) {
  union { float f; uint32_t u; } c; c.f = f;
  uint32_t u = c.u;
  uint32_t r = (u + 0x7fffu + ((u >> 16) & 1u)) >> 16;  // RNE
  union { uint16_t s; bf16_t b; } o; o.s = (uint16_t)r;
  return o.b;
}

// Load one 16x32 bf16 fragment (A-style layout; B uses same with lane<->N).
// lane<16: rows 0..15, K {0..7}U{16..23}; lane>=16: K {8..15}U{24..31}.
// Caller guarantees rows in-bounds (all dims padded). PF: speculative
// prefetch 4 k-tiles ahead (dropped by HW if address invalid).
template <bool PF>
__device__ __forceinline__ v16bf load_frag(const bf16_t* base, long ld,
                                           int row0, int k0) {
  int lane = threadIdx.x & 31;
  int r  = row0 + (lane & 15);
  int kb = k0 + ((lane & 16) ? 8 : 0);
  const bf16_t* p = base + (long)r * ld + kb;
  union { v16bf v; v8bf h[2]; } u;
  u.h[0] = *(const v8bf*)(p);
  u.h[1] = *(const v8bf*)(p + 16);
  if (PF) __builtin_prefetch((const void*)(p + 128), 0, 1);
  return u.v;
}

__device__ __forceinline__ void load_set(const bf16_t* A, long lda, int m0,
                                         const bf16_t* Bm, long ldb, int n0,
                                         int k0, v16bf a[4], v16bf b[2]) {
  a[0] = load_frag<false>(A, lda, m0,      k0);
  a[1] = load_frag<false>(A, lda, m0 + 16, k0);
  a[2] = load_frag<false>(A, lda, m0 + 32, k0);
  a[3] = load_frag<false>(A, lda, m0 + 48, k0);
  b[0] = load_frag<true >(Bm, ldb, n0,      k0);
  b[1] = load_frag<true >(Bm, ldb, n0 + 16, k0);
}

__device__ __forceinline__ void mma_set(v8f acc[4][2], const v16bf a[4],
                                        const v16bf b[2]) {
#pragma unroll
  for (int i = 0; i < 4; ++i) {
    acc[i][0] = __builtin_amdgcn_wmma_f32_16x16x32_bf16(
        false, a[i], false, b[0], (short)0, acc[i][0], false, false);
    acc[i][1] = __builtin_amdgcn_wmma_f32_16x16x32_bf16(
        false, a[i], false, b[1], (short)0, acc[i][1], false, false);
  }
}

// ---------------- generic batched WMMA GEMM:  C = A[M,K] * B[N,K]^T ----------------
// Block: 256 thr = 8 waves (4 along M x 2 along N); block tile 256x64,
// wave tile 64x32 (MI=4 x NI=2 of 16x16), K double-buffered in steps of 64.
// Caller guarantees M % 256 == 0, N % 64 == 0, K % 64 == 0 (all padded).
// MODE 0: bf16 out   MODE 1: f32 out   MODE 2: f32 out = R + acc (residual)
// MODE 3: attention scores: acc*scale - (m-n)*slope(h), causal -> NEG, f32 out
// batch base offset = (z / nH)*Out + (z % nH)*In  for A, B, C.
template <int MODE>
__global__ void __launch_bounds__(256)
gemm_wmma(const bf16_t* A, long lda, long aOut, long aIn,
          const bf16_t* Bm, long ldb, long bOut, long bIn,
          void* Cv, long ldc, long cOut, long cIn,
          const float* R, long ldr,
          int Mr, int Nr, int Kr, int nH, float scale) {
  int z  = blockIdx.z;
  int zo = z / nH, zi = z - zo * nH;
  A  += (long)zo * aOut + (long)zi * aIn;
  Bm += (long)zo * bOut + (long)zi * bIn;
  long cofs = (long)zo * cOut + (long)zi * cIn;

  int wid   = threadIdx.x >> 5;   // 8 waves
  int waveM = wid >> 1;           // 0..3
  int waveN = wid & 1;            // 0..1
  int m0 = blockIdx.y * 256 + waveM * 64;
  int n0 = blockIdx.x * 64  + waveN * 32;

  v8f acc[4][2];
#pragma unroll
  for (int i = 0; i < 4; ++i)
#pragma unroll
    for (int j = 0; j < 2; ++j) {
      v8f zv = {0.f, 0.f, 0.f, 0.f, 0.f, 0.f, 0.f, 0.f};
      acc[i][j] = zv;
    }

  // scores fast path: whole block strictly above the diagonal -> all NEG
  bool allmask = (MODE == 3) &&
                 ((int)blockIdx.x * 64 > (int)blockIdx.y * 256 + 255);

  if (!allmask) {
    v16bf aA[4], bA[2], aB[4], bB[2];
    load_set(A, lda, m0, Bm, ldb, n0, 0, aA, bA);        // stage 0
    for (int k0 = 0; k0 < Kr; k0 += 64) {
      load_set(A, lda, m0, Bm, ldb, n0, k0 + 32, aB, bB); // stage 1 in flight
      mma_set(acc, aA, bA);                               // compute stage 0
      if (k0 + 64 < Kr)
        load_set(A, lda, m0, Bm, ldb, n0, k0 + 64, aA, bA); // stage 0 refill
      mma_set(acc, aB, bB);                               // compute stage 1
    }
  }

  int lane = threadIdx.x & 31;
  int rsub = (lane >> 4) << 3;   // 0 or 8
  int csub = lane & 15;
  float slope = 0.f;
  if (MODE == 3) slope = exp2f(-(8.0f / (float)nH) * (float)(zi + 1));

#pragma unroll
  for (int i = 0; i < 4; ++i)
#pragma unroll
    for (int j = 0; j < 2; ++j)
#pragma unroll
      for (int e = 0; e < 8; ++e) {
        int m = m0 + i * 16 + rsub + e;
        int n = n0 + j * 16 + csub;
        float v = acc[i][j][e];
        long idx = cofs + (long)m * ldc + n;
        if (MODE == 0) {
          ((bf16_t*)Cv)[idx] = f2bf(v);
        } else if (MODE == 1) {
          ((float*)Cv)[idx] = v;
        } else if (MODE == 2) {
          ((float*)Cv)[idx] = R[(long)m * ldr + n] + v;
        } else {
          float val = (n > m) ? NEGV
                              : (v * scale - (float)(m - n) * slope);
          ((float*)Cv)[idx] = val;
        }
      }
}

// ---------------- elementwise / reduction kernels ----------------
__global__ void __launch_bounds__(256)
embed_kernel(const int* ids, const float* emb, float* x, long total, int D) {
  long stride = (long)gridDim.x * 256;
  for (long i = (long)blockIdx.x * 256 + threadIdx.x; i < total; i += stride) {
    long row = i / D;
    int  col = (int)(i - row * D);
    x[i] = emb[(long)ids[row] * D + col];
  }
}

// h = x / (mean|x| + eps) * w  -> bf16   (one block per row)
__global__ void __launch_bounds__(256)
menorm_kernel(const float* x, const float* w, bf16_t* out, int D) {
  __shared__ float red[256];
  long row = blockIdx.x;
  const float* xr = x + row * (long)D;
  float s = 0.f;
  for (int c = threadIdx.x; c < D; c += 256) s += fabsf(xr[c]);
  red[threadIdx.x] = s;
  __syncthreads();
  for (int st = 128; st > 0; st >>= 1) {
    if ((int)threadIdx.x < st) red[threadIdx.x] += red[threadIdx.x + st];
    __syncthreads();
  }
  float inv = 1.0f / (red[0] / (float)D + EPSV);
  bf16_t* orow = out + row * (long)D;
  for (int c = threadIdx.x; c < D; c += 256) orow[c] = f2bf(xr[c] * inv * w[c]);
}

// rational softmax over rows of length T=1024 (4 elems/thread), probs -> bf16
__global__ void __launch_bounds__(256)
rsoftmax_kernel(const float* scores, bf16_t* probs) {
  __shared__ float red[256];
  long row = blockIdx.x;
  const float* sr = scores + row * (long)TT;
  bf16_t* pr = probs + row * (long)TT;
  float v[4];
  float mx = -3.4e38f;
#pragma unroll
  for (int q = 0; q < 4; ++q) {
    v[q] = sr[threadIdx.x + q * 256];
    mx = fmaxf(mx, v[q]);
  }
  red[threadIdx.x] = mx;
  __syncthreads();
  for (int st = 128; st > 0; st >>= 1) {
    if ((int)threadIdx.x < st)
      red[threadIdx.x] = fmaxf(red[threadIdx.x], red[threadIdx.x + st]);
    __syncthreads();
  }
  float rowmax = red[0];
  __syncthreads();
  float s = 0.f;
#pragma unroll
  for (int q = 0; q < 4; ++q) {
    float d = v[q] - rowmax;
    float num = 1.0f / (1.0f - d + 0.5f * d * d);
    v[q] = num;
    s += num;
  }
  red[threadIdx.x] = s;
  __syncthreads();
  for (int st = 128; st > 0; st >>= 1) {
    if ((int)threadIdx.x < st) red[threadIdx.x] += red[threadIdx.x + st];
    __syncthreads();
  }
  float inv = 1.0f / (red[0] + EPSV);
#pragma unroll
  for (int q = 0; q < 4; ++q) pr[threadIdx.x + q * 256] = f2bf(v[q] * inv);
}

// vt[b,h,d,t] = V[b,h,t,d]  (V lives inside qkv at col offset 2D)
__global__ void __launch_bounds__(256)
vtrans_kernel(const bf16_t* qkv, bf16_t* vt, long total) {
  long stride = (long)gridDim.x * 256;
  for (long i = (long)blockIdx.x * 256 + threadIdx.x; i < total; i += stride) {
    long t = i % TT;
    long rest = i / TT;
    long d = rest % DHH; rest /= DHH;
    long h = rest % HH;
    long b = rest / HH;
    vt[i] = qkv[(b * TT + t) * (3L * DD) + 2L * DD + h * (long)DHH + d];
  }
}

// act = gate * sigma(gate) * val, padded to FPAD columns with zeros
// gv rows have leading dimension GPAD (cols >= 2F are zero / ignored)
__global__ void __launch_bounds__(256)
gating_kernel(const float* gv, bf16_t* act, long total) {
  long stride = (long)gridDim.x * 256;
  for (long i = (long)blockIdx.x * 256 + threadIdx.x; i < total; i += stride) {
    long m = i / FPAD;
    int  f = (int)(i - m * FPAD);
    float o = 0.f;
    if (f < FF) {
      float g  = gv[m * (long)GPAD + f];
      float vv = gv[m * (long)GPAD + FF + f];
      float sig = 0.5f * (g / (fabsf(g) + 1.0f) + 1.0f);
      o = g * sig * vv;
    }
    act[i] = f2bf(o);
  }
}

// fp32 -> bf16 weight convert with row and column zero-padding.
// src: nmat matrices of [srcRows x lds] (contiguous); dst: [dstRows x ldd].
__global__ void __launch_bounds__(256)
cvt_kernel(const float* src, bf16_t* dst, long nmat, long srcRows, long dstRows,
           long Ks, long lds, long ldd) {
  long per = dstRows * ldd;
  long total = nmat * per;
  long stride = (long)gridDim.x * 256;
  for (long i = (long)blockIdx.x * 256 + threadIdx.x; i < total; i += stride) {
    long mat = i / per;
    long rem = i - mat * per;
    long r = rem / ldd;
    long c = rem - r * ldd;
    float v = 0.f;
    if (r < srcRows && c < Ks) v = src[(mat * srcRows + r) * lds + c];
    dst[i] = f2bf(v);
  }
}

// ---------------- host launcher ----------------
static inline unsigned cdivu(long a, long b) { return (unsigned)((a + b - 1) / b); }

extern "C" void kernel_launch(void* const* d_in, const int* in_sizes, int n_in,
                              void* d_out, int out_size, void* d_ws, size_t ws_size,
                              hipStream_t stream) {
  (void)in_sizes; (void)n_in; (void)out_size; (void)ws_size;
  const int*   ids     = (const int*)  d_in[0];
  const float* embed_w = (const float*)d_in[1];
  const float* qkv_w   = (const float*)d_in[2];
  const float* out_w   = (const float*)d_in[3];
  const float* gate_w  = (const float*)d_in[4];
  const float* wout_w  = (const float*)d_in[5];
  const float* n1_w    = (const float*)d_in[6];
  const float* n2_w    = (const float*)d_in[7];
  const float* nf_w    = (const float*)d_in[8];
  float* logits = (float*)d_out;

  char* ws = (char*)d_ws;
  size_t off = 0;
  auto alloc = [&](size_t bytes) -> char* {
    char* p = ws + off;
    off = (off + bytes + 255) & ~(size_t)255;
    return p;
  };

  // bf16 weights (padded where needed)
  bf16_t* wqkv_bf = (bf16_t*)alloc((size_t)LL * 3 * DD * DD * 2);
  bf16_t* wo_bf   = (bf16_t*)alloc((size_t)LL * DD * DD * 2);
  bf16_t* wg_bf   = (bf16_t*)alloc((size_t)LL * GPAD * DD * 2);
  bf16_t* ww_bf   = (bf16_t*)alloc((size_t)LL * DD * FPAD * 2);
  bf16_t* wemb_bf = (bf16_t*)alloc((size_t)VV * DD * 2);
  // activations
  float*  x       = (float*) alloc((size_t)MMr * DD * 4);
  bf16_t* h_bf    = (bf16_t*)alloc((size_t)MMr * DD * 2);
  bf16_t* qkv_bf  = (bf16_t*)alloc((size_t)MMr * 3 * DD * 2);
  float*  scores  = (float*) alloc((size_t)BB * HH * TT * TT * 4);
  bf16_t* probs   = (bf16_t*)alloc((size_t)BB * HH * TT * TT * 2);
  bf16_t* vt_bf   = (bf16_t*)alloc((size_t)BB * HH * DHH * TT * 2);
  bf16_t* o_bf    = (bf16_t*)alloc((size_t)MMr * DD * 2);
  float*  gv      = (float*) alloc((size_t)MMr * GPAD * 4);
  bf16_t* act_bf  = (bf16_t*)alloc((size_t)MMr * FPAD * 2);

  dim3 blk(256);

  // ---- weight conversion (once per launch) ----
  cvt_kernel<<<cdivu((long)LL * 3 * DD * DD, 256), blk, 0, stream>>>(
      qkv_w, wqkv_bf, LL, 3 * DD, 3 * DD, DD, DD, DD);
  cvt_kernel<<<cdivu((long)LL * DD * DD, 256), blk, 0, stream>>>(
      out_w, wo_bf, LL, DD, DD, DD, DD, DD);
  cvt_kernel<<<cdivu((long)LL * GPAD * DD, 256), blk, 0, stream>>>(
      gate_w, wg_bf, LL, 2 * FF, GPAD, DD, DD, DD);
  cvt_kernel<<<cdivu((long)LL * DD * FPAD, 256), blk, 0, stream>>>(
      wout_w, ww_bf, LL, DD, DD, FF, FF, FPAD);
  cvt_kernel<<<cdivu((long)VV * DD, 256), blk, 0, stream>>>(
      embed_w, wemb_bf, 1, VV, VV, DD, DD, DD);

  // ---- embedding ----
  embed_kernel<<<cdivu((long)MMr * DD, 256), blk, 0, stream>>>(
      ids, embed_w, x, (long)MMr * DD, DD);

  for (int l = 0; l < LL; ++l) {
    const bf16_t* wqkv_l = wqkv_bf + (size_t)l * 3 * DD * DD;
    const bf16_t* wo_l   = wo_bf   + (size_t)l * DD * DD;
    const bf16_t* wg_l   = wg_bf   + (size_t)l * GPAD * DD;
    const bf16_t* ww_l   = ww_bf   + (size_t)l * DD * FPAD;

    // norm1 -> h
    menorm_kernel<<<MMr, blk, 0, stream>>>(x, n1_w + (size_t)l * DD, h_bf, DD);

    // qkv = h @ Wqkv^T   [2048 x 3072]
    gemm_wmma<0><<<dim3(3 * DD / 64, MMr / 256, 1), blk, 0, stream>>>(
        h_bf, DD, 0, 0, wqkv_l, DD, 0, 0, qkv_bf, 3 * DD, 0, 0,
        nullptr, 0, MMr, 3 * DD, DD, 1, 1.0f);

    // scores[b,h] = Q K^T * SCALE + alibi, causal NEG    (z = b*H + h)
    gemm_wmma<3><<<dim3(TT / 64, TT / 256, BB * HH), blk, 0, stream>>>(
        qkv_bf,      3 * DD, (long)TT * 3 * DD, DHH,
        qkv_bf + DD, 3 * DD, (long)TT * 3 * DD, DHH,
        scores, TT, (long)HH * TT * TT, (long)TT * TT,
        nullptr, 0, TT, TT, DHH, HH, 0.125f);

    // rational softmax rows -> probs (bf16)
    rsoftmax_kernel<<<BB * HH * TT, blk, 0, stream>>>(scores, probs);

    // transpose V
    vtrans_kernel<<<cdivu((long)BB * HH * DHH * TT, 256), blk, 0, stream>>>(
        qkv_bf, vt_bf, (long)BB * HH * DHH * TT);

    // O[b,h] = P @ V -> o_bf laid out [B,T,H,DH] = [M,D]
    gemm_wmma<0><<<dim3(DHH / 64, TT / 256, BB * HH), blk, 0, stream>>>(
        probs, TT, (long)HH * TT * TT, (long)TT * TT,
        vt_bf, TT, (long)HH * DHH * TT, (long)DHH * TT,
        o_bf, DD, (long)TT * DD, DHH,
        nullptr, 0, TT, DHH, TT, HH, 1.0f);

    // x = x + o @ Wout^T
    gemm_wmma<2><<<dim3(DD / 64, MMr / 256, 1), blk, 0, stream>>>(
        o_bf, DD, 0, 0, wo_l, DD, 0, 0, x, DD, 0, 0,
        x, DD, MMr, DD, DD, 1, 1.0f);

    // norm2 -> h
    menorm_kernel<<<MMr, blk, 0, stream>>>(x, n2_w + (size_t)l * DD, h_bf, DD);

    // gv = h @ Wgate^T   [2048 x GPAD] (weight rows zero-padded 5460->5504)
    gemm_wmma<1><<<dim3(GPAD / 64, MMr / 256, 1), blk, 0, stream>>>(
        h_bf, DD, 0, 0, wg_l, DD, 0, 0, gv, GPAD, 0, 0,
        nullptr, 0, MMr, GPAD, DD, 1, 1.0f);

    // gating -> act (bf16, K padded)
    gating_kernel<<<cdivu((long)MMr * FPAD, 256), blk, 0, stream>>>(
        gv, act_bf, (long)MMr * FPAD);

    // x = x + act @ Wwout^T  (K = FPAD, zero-padded)
    gemm_wmma<2><<<dim3(DD / 64, MMr / 256, 1), blk, 0, stream>>>(
        act_bf, FPAD, 0, 0, ww_l, FPAD, 0, 0, x, DD, 0, 0,
        x, DD, MMr, DD, FPAD, 1, 1.0f);
  }

  // final norm -> h ; logits = h @ embed^T
  menorm_kernel<<<MMr, blk, 0, stream>>>(x, nf_w, h_bf, DD);
  gemm_wmma<1><<<dim3(VV / 64, MMr / 256, 1), blk, 0, stream>>>(
      h_bf, DD, 0, 0, wemb_bf, DD, 0, 0, logits, VV, 0, 0,
      nullptr, 0, MMr, VV, DD, 1, 1.0f);
}
